// TransformerDecoder_22582938043109
// MI455X (gfx1250) — matrix operand order repaired and checked by reading, not compiled, not run on hardware
//
#include <hip/hip_runtime.h>
#include <hip/hip_bf16.h>

// ---------------------------------------------------------------------------
// Transformer decoder layer for MI455X (gfx1250, wave32, WMMA bf16).
// B=2, S=2048, D=2048, H=16, DH=128, DFF=8192.
// GEMMs: double-buffered LDS fed by global_load_async_to_lds_b128 (ASYNCcnt),
// compute on v_wmma_f32_16x16x32_bf16. Attention: flash-style, same async
// staging for K/V tiles.  Workspace: ~470 MB.
// ---------------------------------------------------------------------------

#define D_   2048
#define S_   2048
#define H_   16
#define DH_  128
#define DFF_ 8192
#define BTOK 4096   // B * S
#define EPS_ 1e-6f
#define ATT_SCALE 0.08838834764831845f   // 1/sqrt(128)

typedef __attribute__((ext_vector_type(16))) __bf16 v16bf;
typedef __attribute__((ext_vector_type(8)))  __bf16 v8bf;
typedef __attribute__((ext_vector_type(8)))  float  v8f;

__device__ __forceinline__ v8f wmma_bf16(v16bf a, v16bf b, v8f c) {
  // D = A(16x32 bf16) x B(32x16 bf16) + C(16x16 f32)
  return __builtin_amdgcn_wmma_f32_16x16x32_bf16(false, a, false, b, (short)0, c,
                                                 false, false);
}

// A-matrix bf16 16x32 fragment: lane holds row = lane%16,
// elements K = kb+0..15 contiguous (lane>=16 -> kb=16). Two 16B LDS reads.
__device__ __forceinline__ v16bf ldsA_frag(const __bf16* p) {
  v8bf lo = *(const v8bf*)p;
  v8bf hi = *(const v8bf*)(p + 8);
  return __builtin_shufflevector(lo, hi, 0, 1, 2, 3, 4, 5, 6, 7,
                                         8, 9, 10, 11, 12, 13, 14, 15);
}

// B-matrix bf16 32x16 fragment: lane holds col = lane%16,
// elements K = kb+0..15 contiguous (lane>=16 -> kb=16). Two 16B LDS reads.
__device__ __forceinline__ v16bf ldsB_frag(const __bf16* p) {
  v8bf lo = *(const v8bf*)p;
  v8bf hi = *(const v8bf*)(p + 8);
  return __builtin_shufflevector(lo, hi, 0, 1, 2, 3, 4, 5, 6, 7,
                                         8, 9, 10, 11, 12, 13, 14, 15);
}

// Generic LDS pointer -> LDS byte offset (ISA: LDS aperture addr[31:0]).
__device__ __forceinline__ unsigned lds_lo32(const void* p) {
  return (unsigned)(unsigned long long)p;
}

// Async DMA: 16B global -> LDS, GVS addressing (SGPR64 base + VGPR32 offset).
// INST_OFFSET applies to both global and LDS sides (ISA 10.x async pseudo).
template <int OFF>
__device__ __forceinline__ void async_b128(unsigned ldsaddr, unsigned voff,
                                           unsigned long long sbase) {
  asm volatile("global_load_async_to_lds_b128 %0, %1, %2 offset:%3"
               :: "v"(ldsaddr), "v"(voff), "s"(sbase), "i"(OFF)
               : "memory");
}

__device__ __forceinline__ void wait_async0() {
  asm volatile("s_wait_asynccnt 0x0" ::: "memory");
}

// ---------------------------------------------------------------------------
// f32 -> bf16 conversion (weights, done every call: no cached state)
// ---------------------------------------------------------------------------
__global__ __launch_bounds__(256) void cvt_f32_to_bf16(const float* __restrict__ a,
                                                       __bf16* __restrict__ b, int n) {
  int stride = gridDim.x * blockDim.x;
  for (int i = blockIdx.x * blockDim.x + threadIdx.x; i < n; i += stride)
    b[i] = (__bf16)a[i];
}

// ---------------------------------------------------------------------------
// RMSNorm (faithful to source quirk: multiply by sqrt(mean(x^2)+eps))
// ---------------------------------------------------------------------------
__global__ __launch_bounds__(256) void rmsnorm_to_bf16(const float* __restrict__ x,
                                                       const float* __restrict__ g,
                                                       __bf16* __restrict__ y) {
  __shared__ float red[256];
  const int row = blockIdx.x;
  const float* xr = x + (size_t)row * D_;
  float s = 0.f;
  for (int i = threadIdx.x; i < D_; i += 256) {
    float v = xr[i];
    s += v * v;
  }
  red[threadIdx.x] = s;
  __syncthreads();
  for (int off = 128; off > 0; off >>= 1) {
    if (threadIdx.x < off) red[threadIdx.x] += red[threadIdx.x + off];
    __syncthreads();
  }
  float scale = sqrtf(red[0] * (1.f / D_) + EPS_);
  __bf16* yr = y + (size_t)row * D_;
  for (int i = threadIdx.x; i < D_; i += 256)
    yr[i] = (__bf16)(xr[i] * scale * g[i]);
}

// ---------------------------------------------------------------------------
// GEMM: C[M,N] = A[M,K](bf16) * B[N,K](bf16)^T  (+bias[col]) (+resid[row,col])
// Block tile 128x128, K-step 64, double-buffered async LDS staging.
// 8 waves; wave tile 32x64 -> 16 WMMA per K-step, one barrier per K-step.
// ---------------------------------------------------------------------------
template <bool OUT_BF16>
__global__ __launch_bounds__(256)
void gemm_bf16_wmma(const __bf16* __restrict__ A, const __bf16* __restrict__ B,
                    const float* __restrict__ bias, const float* __restrict__ resid,
                    void* __restrict__ Cout, int M, int N, int K) {
  __shared__ __bf16 sa[2][128][64];
  __shared__ __bf16 sb[2][128][64];
  const int tid  = threadIdx.x;
  const int lane = tid & 31;
  const int wid  = tid >> 5;
  const int wm0  = (wid & 3) * 32;   // 4 waves along M
  const int wn0  = (wid >> 2) * 64;  // 2 waves along N
  const int blkM = blockIdx.y * 128;
  const int blkN = blockIdx.x * 128;

  v8f acc[2][4] = {};

  // Staging: 256 threads, each owns a 64B (32-bf16) chunk of a 128x64 tile.
  const int lr  = tid >> 1;          // tile row 0..127
  const int lc  = (tid & 1) * 32;    // col half (elements)
  const unsigned voffA = (unsigned)(((size_t)lr * K + lc) * sizeof(__bf16));
  const unsigned voffB = voffA;      // B tile has identical geometry
  const unsigned long long aBase =
      (unsigned long long)(A + (size_t)blkM * K);
  const unsigned long long bBase =
      (unsigned long long)(B + (size_t)blkN * K);
  unsigned ldsA[2], ldsB[2];
  ldsA[0] = lds_lo32(&sa[0][lr][lc]);
  ldsA[1] = lds_lo32(&sa[1][lr][lc]);
  ldsB[0] = lds_lo32(&sb[0][lr][lc]);
  ldsB[1] = lds_lo32(&sb[1][lr][lc]);

  auto stage = [&](int p, int kb) {
    unsigned long long ab = aBase + (size_t)kb * sizeof(__bf16);
    unsigned long long bb = bBase + (size_t)kb * sizeof(__bf16);
    async_b128<0>(ldsA[p], voffA, ab);
    async_b128<16>(ldsA[p], voffA, ab);
    async_b128<32>(ldsA[p], voffA, ab);
    async_b128<48>(ldsA[p], voffA, ab);
    async_b128<0>(ldsB[p], voffB, bb);
    async_b128<16>(ldsB[p], voffB, bb);
    async_b128<32>(ldsB[p], voffB, bb);
    async_b128<48>(ldsB[p], voffB, bb);
  };

  const int akb = (lane >> 4) * 16;  // A frag K base
  const int bkb = (lane >> 4) * 16;  // B frag K base
  const int l15 = lane & 15;

  stage(0, 0);
  int p = 0;
  for (int k0 = 0; k0 < K; k0 += 64) {
    wait_async0();        // own async stores to buffer p complete
    __syncthreads();      // everyone's stores complete; buffer p^1 free
    if (k0 + 64 < K) stage(p ^ 1, k0 + 64);  // DMA next tile during compute

#pragma unroll
    for (int ks = 0; ks < 2; ++ks) {
      const int kb = ks * 32;
      v16bf af[2], bfv[4];
#pragma unroll
      for (int mt = 0; mt < 2; ++mt)
        af[mt] = ldsA_frag(&sa[p][wm0 + mt * 16 + l15][kb + akb]);
#pragma unroll
      for (int nt = 0; nt < 4; ++nt)
        bfv[nt] = ldsB_frag(&sb[p][wn0 + nt * 16 + l15][kb + bkb]);
#pragma unroll
      for (int mt = 0; mt < 2; ++mt)
#pragma unroll
        for (int nt = 0; nt < 4; ++nt)
          acc[mt][nt] = wmma_bf16(af[mt], bfv[nt], acc[mt][nt]);
    }
    p ^= 1;
  }

  // Epilogue. C tile layout: lane<16 -> M=i, lane>=16 -> M=i+8; N = lane%16.
  const int mofs = (lane >> 4) * 8;
#pragma unroll
  for (int mt = 0; mt < 2; ++mt) {
#pragma unroll
    for (int nt = 0; nt < 4; ++nt) {
      int col = blkN + wn0 + nt * 16 + l15;
      float bv = bias ? bias[col] : 0.f;
#pragma unroll
      for (int i = 0; i < 8; ++i) {
        int row = blkM + wm0 + mt * 16 + i + mofs;
        size_t idx = (size_t)row * N + col;
        float v = acc[mt][nt][i] + bv;
        if (resid) v += resid[idx];
        if (OUT_BF16)
          ((__bf16*)Cout)[idx] = (__bf16)v;
        else
          ((float*)Cout)[idx] = v;
      }
    }
  }
}

// ---------------------------------------------------------------------------
// Flash-style attention. Grid: (S/16, H, B). Block: 128 threads (4 waves).
// Two-pass online softmax; K/V tiles staged with async DMA.
// Output stored TRANSPOSED (scr[b][h*128+dh][s]) which reproduces the
// reference's transpose(0,1,3,2).reshape(B,S,D) scramble exactly (S == D).
// ---------------------------------------------------------------------------
__global__ __launch_bounds__(128)
void flash_attention(const __bf16* __restrict__ q, const __bf16* __restrict__ k,
                     const __bf16* __restrict__ v, __bf16* __restrict__ scr) {
  __shared__ __bf16 ldsQ[16][128];
  __shared__ __bf16 ldsK[64][128];
  __shared__ __bf16 ldsV[64][128];
  __shared__ float  ldsS[16][64];
  __shared__ __bf16 ldsP[16][64];
  __shared__ float  ldsM[16];
  __shared__ float  ldsL[16];

  const int tid  = threadIdx.x;
  const int lane = tid & 31;
  const int wid  = tid >> 5;
  const int l15  = lane & 15;
  const int mofs = (lane >> 4) * 8;

  const int q0 = blockIdx.x * 16;
  const int h  = blockIdx.y;
  const int b  = blockIdx.z;

  const __bf16* qg = q + (size_t)b * S_ * D_ + h * DH_;
  const __bf16* kg = k + (size_t)b * S_ * D_ + h * DH_;
  const __bf16* vg = v + (size_t)b * S_ * D_ + h * DH_;
  __bf16* scrg = scr + (size_t)b * D_ * S_;

  // Staging geometry for 64x128 K/V tiles: 128 threads, 128B per thread.
  const int sr   = tid >> 1;            // tile row 0..63
  const int sc   = (tid & 1) * 64;      // col half (elements)
  const unsigned voffKV = (unsigned)(((size_t)sr * D_ + sc) * sizeof(__bf16));
  const unsigned ldsKa = lds_lo32(&ldsK[sr][sc]);
  const unsigned ldsVa = lds_lo32(&ldsV[sr][sc]);

  auto stage_tile = [&](unsigned ldsaddr, const __bf16* gbase, int kt) {
    unsigned long long sb =
        (unsigned long long)(gbase + (size_t)kt * 64 * D_);
    async_b128<0>(ldsaddr, voffKV, sb);
    async_b128<16>(ldsaddr, voffKV, sb);
    async_b128<32>(ldsaddr, voffKV, sb);
    async_b128<48>(ldsaddr, voffKV, sb);
    async_b128<64>(ldsaddr, voffKV, sb);
    async_b128<80>(ldsaddr, voffKV, sb);
    async_b128<96>(ldsaddr, voffKV, sb);
    async_b128<112>(ldsaddr, voffKV, sb);
  };

  // Stage Q tile [16 x 128] once; build per-wave A fragments (live in VGPRs).
  {
    int r = tid >> 3, seg = tid & 7;  // 16 rows x 8 segs of 16 bf16
    const uint4* src = (const uint4*)(qg + (size_t)(q0 + r) * D_ + seg * 16);
    uint4* dst = (uint4*)&ldsQ[r][seg * 16];
    dst[0] = src[0];
    dst[1] = src[1];
  }
  if (tid < 16) { ldsM[tid] = -1e30f; ldsL[tid] = 0.f; }
  __syncthreads();

  v16bf qfrag[4];
#pragma unroll
  for (int kk = 0; kk < 4; ++kk)
    qfrag[kk] = ldsA_frag(&ldsQ[l15][kk * 32 + (lane >> 4) * 16]);

  // Computes scaled scores for key tile kt into ldsS (each wave -> 16 keys).
  auto tile_scores = [&](int kt) {
    stage_tile(ldsKa, kg, kt);
    wait_async0();
    __syncthreads();
    const int n0 = wid * 16;
    v16bf bfv[4];
#pragma unroll
    for (int kk = 0; kk < 4; ++kk)
      bfv[kk] = ldsB_frag(&ldsK[n0 + l15][kk * 32 + (lane >> 4) * 16]);
    v8f s = {};
#pragma unroll
    for (int kk = 0; kk < 4; ++kk) s = wmma_bf16(qfrag[kk], bfv[kk], s);
#pragma unroll
    for (int i = 0; i < 8; ++i)
      ldsS[i + mofs][n0 + l15] = s[i] * ATT_SCALE;
    __syncthreads();
  };

  // ---------------- Pass 1: running row max / denom over all 2048 keys ----
  for (int kt = 0; kt < S_ / 64; ++kt) {
    tile_scores(kt);
    if (tid < 16) {
      float mOld = ldsM[tid], lOld = ldsL[tid];
      float tmax = -1e30f;
#pragma unroll 8
      for (int j = 0; j < 64; ++j) tmax = fmaxf(tmax, ldsS[tid][j]);
      float mNew = fmaxf(mOld, tmax);
      float l = lOld * __expf(mOld - mNew);
#pragma unroll 8
      for (int j = 0; j < 64; ++j) l += __expf(ldsS[tid][j] - mNew);
      ldsM[tid] = mNew;
      ldsL[tid] = l;
    }
    __syncthreads();
  }

  // ---------------- Pass 2: recompute P, accumulate P @ V -----------------
  v8f acc[2] = {};  // wave owns dh columns [wid*32, wid*32+32)
  for (int kt = 0; kt < S_ / 64; ++kt) {
    tile_scores(kt);
    // P = exp(s - m[row]) as bf16; stage V tile with async DMA meanwhile.
    stage_tile(ldsVa, vg, kt);
#pragma unroll
    for (int j = 0; j < 8; ++j) {
      int e = tid * 8 + j;
      int rr = e >> 6, cc = e & 63;
      ldsP[rr][cc] = (__bf16)__expf(ldsS[rr][cc] - ldsM[rr]);
    }
    wait_async0();
    __syncthreads();
#pragma unroll
    for (int ks = 0; ks < 2; ++ks) {
      v16bf af = ldsA_frag(&ldsP[l15][ks * 32 + (lane >> 4) * 16]);
      const int kb = ks * 32 + (lane >> 4) * 16;
      v16bf bfv[2];
#pragma unroll
      for (int nt = 0; nt < 2; ++nt) {
        int dhcol = wid * 32 + nt * 16 + l15;
#pragma unroll
        for (int j = 0; j < 16; ++j) bfv[nt][j] = ldsV[kb + j][dhcol];
      }
#pragma unroll
      for (int nt = 0; nt < 2; ++nt) acc[nt] = wmma_bf16(af, bfv[nt], acc[nt]);
    }
    __syncthreads();
  }

  // Epilogue: divide by softmax denom, store transposed: scr[h*128+dh][s].
#pragma unroll
  for (int nt = 0; nt < 2; ++nt) {
    int dhcol = wid * 32 + nt * 16 + l15;
#pragma unroll
    for (int i = 0; i < 8; ++i) {
      int m = i + mofs;
      float val = acc[nt][i] / ldsL[m];
      scrg[(size_t)(h * DH_ + dhcol) * S_ + q0 + m] = (__bf16)val;
    }
  }
}

// ---------------------------------------------------------------------------
// Gated FFN elementwise: inter = relu(gate) * up
// ---------------------------------------------------------------------------
__global__ __launch_bounds__(256) void glu_mul(const __bf16* __restrict__ gate,
                                               const __bf16* __restrict__ up,
                                               __bf16* __restrict__ out, int n) {
  int stride = gridDim.x * blockDim.x;
  for (int i = blockIdx.x * blockDim.x + threadIdx.x; i < n; i += stride) {
    float gv = (float)gate[i];
    float uv = (float)up[i];
    out[i] = (__bf16)(fmaxf(gv, 0.f) * uv);
  }
}

// ---------------------------------------------------------------------------
extern "C" void kernel_launch(void* const* d_in, const int* in_sizes, int n_in,
                              void* d_out, int out_size, void* d_ws, size_t ws_size,
                              hipStream_t stream) {
  const float* x      = (const float*)d_in[0];
  const float* wq     = (const float*)d_in[1];
  const float* bq     = (const float*)d_in[2];
  const float* wk     = (const float*)d_in[3];
  const float* bk     = (const float*)d_in[4];
  const float* wv     = (const float*)d_in[5];
  const float* bv     = (const float*)d_in[6];
  const float* wo     = (const float*)d_in[7];
  const float* bo     = (const float*)d_in[8];
  const float* w_gate = (const float*)d_in[9];
  const float* w_up   = (const float*)d_in[10];
  const float* w_down = (const float*)d_in[11];
  const float* g_att  = (const float*)d_in[12];
  const float* g_ffn  = (const float*)d_in[13];

  char* ws = (char*)d_ws;
  auto bump = [&](size_t bytes) -> char* {
    char* p = ws;
    ws += (bytes + 255) & ~(size_t)255;
    return p;
  };

  const size_t nDD  = (size_t)D_ * D_;       // 4.19M
  const size_t nFD  = (size_t)DFF_ * D_;     // 16.8M
  const size_t nTok = (size_t)BTOK * D_;     // 8.39M
  const size_t nInt = (size_t)BTOK * DFF_;   // 33.6M

  __bf16* wq_bf = (__bf16*)bump(nDD * 2);
  __bf16* wk_bf = (__bf16*)bump(nDD * 2);
  __bf16* wv_bf = (__bf16*)bump(nDD * 2);
  __bf16* wo_bf = (__bf16*)bump(nDD * 2);
  __bf16* wg_bf = (__bf16*)bump(nFD * 2);
  __bf16* wu_bf = (__bf16*)bump(nFD * 2);
  __bf16* wd_bf = (__bf16*)bump(nFD * 2);
  __bf16* xn_bf = (__bf16*)bump(nTok * 2);
  __bf16* q_bf  = (__bf16*)bump(nTok * 2);
  __bf16* k_bf  = (__bf16*)bump(nTok * 2);
  __bf16* v_bf  = (__bf16*)bump(nTok * 2);
  __bf16* scr_bf = (__bf16*)bump(nTok * 2);
  float*  h_f32 = (float*)bump(nTok * 4);
  __bf16* hn_bf = (__bf16*)bump(nTok * 2);
  __bf16* gate_bf = (__bf16*)bump(nInt * 2);
  __bf16* up_bf   = (__bf16*)bump(nInt * 2);
  __bf16* inter_bf = (__bf16*)bump(nInt * 2);
  (void)ws_size; (void)n_in; (void)in_sizes; (void)out_size;

  // Weight conversion f32 -> bf16 (bf16 copies fit in 192MB L2 for GEMMs).
  cvt_f32_to_bf16<<<2048, 256, 0, stream>>>(wq, wq_bf, (int)nDD);
  cvt_f32_to_bf16<<<2048, 256, 0, stream>>>(wk, wk_bf, (int)nDD);
  cvt_f32_to_bf16<<<2048, 256, 0, stream>>>(wv, wv_bf, (int)nDD);
  cvt_f32_to_bf16<<<2048, 256, 0, stream>>>(wo, wo_bf, (int)nDD);
  cvt_f32_to_bf16<<<4096, 256, 0, stream>>>(w_gate, wg_bf, (int)nFD);
  cvt_f32_to_bf16<<<4096, 256, 0, stream>>>(w_up, wu_bf, (int)nFD);
  cvt_f32_to_bf16<<<4096, 256, 0, stream>>>(w_down, wd_bf, (int)nFD);

  // xn = rmsnorm(x, g_att)
  rmsnorm_to_bf16<<<BTOK, 256, 0, stream>>>(x, g_att, xn_bf);

  // Q/K/V projections
  dim3 gDD(D_ / 128, BTOK / 128);  // (16, 32)
  gemm_bf16_wmma<true><<<gDD, 256, 0, stream>>>(xn_bf, wq_bf, bq, nullptr, q_bf,
                                                BTOK, D_, D_);
  gemm_bf16_wmma<true><<<gDD, 256, 0, stream>>>(xn_bf, wk_bf, bk, nullptr, k_bf,
                                                BTOK, D_, D_);
  gemm_bf16_wmma<true><<<gDD, 256, 0, stream>>>(xn_bf, wv_bf, bv, nullptr, v_bf,
                                                BTOK, D_, D_);

  // Attention (writes transposed per-batch matrix = reference's scramble)
  flash_attention<<<dim3(S_ / 16, H_, 2), 128, 0, stream>>>(q_bf, k_bf, v_bf,
                                                            scr_bf);

  // h = x + scr @ wo^T + bo
  gemm_bf16_wmma<false><<<gDD, 256, 0, stream>>>(scr_bf, wo_bf, bo, x, h_f32,
                                                 BTOK, D_, D_);

  // hn = rmsnorm(h, g_ffn)
  rmsnorm_to_bf16<<<BTOK, 256, 0, stream>>>(h_f32, g_ffn, hn_bf);

  // FFN gate/up
  dim3 gFD(DFF_ / 128, BTOK / 128);  // (64, 32)
  gemm_bf16_wmma<true><<<gFD, 256, 0, stream>>>(hn_bf, wg_bf, nullptr, nullptr,
                                                gate_bf, BTOK, DFF_, D_);
  gemm_bf16_wmma<true><<<gFD, 256, 0, stream>>>(hn_bf, wu_bf, nullptr, nullptr,
                                                up_bf, BTOK, DFF_, D_);
  glu_mul<<<8192, 256, 0, stream>>>(gate_bf, up_bf, inter_bf, (int)nInt);

  // out = h + inter @ w_down^T
  gemm_bf16_wmma<false><<<gDD, 256, 0, stream>>>(inter_bf, wd_bf, nullptr, h_f32,
                                                 (float*)d_out, BTOK, D_, DFF_);
}
